// DynamicRegimeGNN_89094801588569
// MI455X (gfx1250) — matrix operand order repaired and checked by reading, not compiled, not run on hardware
//
#include <hip/hip_runtime.h>
#include <math.h>

#define T_  30
#define NT_ 2000
#define B_  4
#define NODES_ 500
#define R_  3
#define E_  64000
#define FIN_ 37
#define F_  128
#define H_  256
#define NB_ 4
#define M_  (T_*NT_)   // 60000 batched rows (T x NT)

typedef __bf16 bf16_t;
typedef __attribute__((ext_vector_type(16))) __bf16 v16bf;
typedef __attribute__((ext_vector_type(8)))  __bf16 v8bf;
typedef __attribute__((ext_vector_type(8)))  float  v8f;

__device__ __forceinline__ bf16_t f2bf(float f) {
  union { float f; unsigned u; } v; v.f = f;
  unsigned r = (v.u + 0x7FFFu + ((v.u >> 16) & 1u)) >> 16;  // RNE truncate
  unsigned short s = (unsigned short)r;
  return __builtin_bit_cast(__bf16, s);
}

__device__ __forceinline__ float eluf(float x) {
  return x > 0.f ? x : (__expf(x) - 1.f);
}

// ---------------------------------------------------------------------------
// WMMA GEMM: C(MxN) = A(MxK) * Bt(NxK)^T,  A/Bt bf16 row-major, C f32.
// Block = 256 threads = 8 waves; each wave computes a 16(M) x 128(N) strip.
// A fragment (16x32 bf16): lane half h, lw=lane&15: row=rowBase+lw,
//   elems 0..15 = K[k0+16h .. +15]                                (2x b128)
// B fragment (32x16 bf16): col = nBase+16j+lw,
//   elems 0..15 = K[k0+16h .. +15]                                (1x 32B)
// ---------------------------------------------------------------------------
template<int KTOT, bool BIAS, bool ELU, bool SF32, bool SBF16>
__global__ __launch_bounds__(256)
void gemm_wmma(const bf16_t* __restrict__ A, const bf16_t* __restrict__ Bt,
               const float* __restrict__ bias, float* __restrict__ Cf,
               bf16_t* __restrict__ Cb, int M, int ldc)
{
  const int wave = threadIdx.x >> 5;
  const int lane = threadIdx.x & 31;
  const int half = lane >> 4;
  const int lw   = lane & 15;
  const int rowBase = blockIdx.x * 128 + wave * 16;
  if (rowBase >= M) return;                 // wave-uniform (M % 16 == 0)
  const int nBase = blockIdx.y * 128;

  v8f acc[8];
#pragma unroll
  for (int j = 0; j < 8; ++j)
#pragma unroll
    for (int i = 0; i < 8; ++i) acc[j][i] = 0.f;

  const bf16_t* arow = A + (size_t)(rowBase + lw) * KTOT;

#pragma unroll
  for (int k0 = 0; k0 < KTOT; k0 += 32) {
    v8bf alo = *reinterpret_cast<const v8bf*>(arow + k0 + half * 16);
    v8bf ahi = *reinterpret_cast<const v8bf*>(arow + k0 + half * 16 + 8);
    v16bf af = __builtin_shufflevector(alo, ahi,
                 0,1,2,3,4,5,6,7,8,9,10,11,12,13,14,15);
#pragma unroll
    for (int j = 0; j < 8; ++j) {
      const bf16_t* brow = Bt + (size_t)(nBase + j * 16 + lw) * KTOT
                              + k0 + half * 16;
      v16bf bfr = *reinterpret_cast<const v16bf*>(brow);
      acc[j] = __builtin_amdgcn_wmma_f32_16x16x32_bf16(
                 false, af, false, bfr, (short)0, acc[j], false, false);
    }
  }

#pragma unroll
  for (int j = 0; j < 8; ++j) {
    const int col = nBase + j * 16 + lw;
#pragma unroll
    for (int i = 0; i < 8; ++i) {
      const int row = rowBase + half * 8 + i;
      float v = acc[j][i];
      if (BIAS) v += bias[col];
      if (ELU)  v = eluf(v);
      if (SF32)  Cf[(size_t)row * ldc + col] = v;
      if (SBF16) Cb[(size_t)row * ldc + col] = f2bf(v);
    }
  }
}

// ---- prep kernels ---------------------------------------------------------

__global__ void k_conv_x(const float* __restrict__ x, bf16_t* __restrict__ xb,
                         int total) {
  int i = blockIdx.x * blockDim.x + threadIdx.x;
  if (i >= total) return;
  int row = i >> 6, k = i & 63;                 // K padded 37 -> 64
  xb[i] = (k < FIN_) ? f2bf(x[row * FIN_ + k]) : f2bf(0.f);
}

// W (K,N) f32 row-major -> Wt (N,Kpad) bf16, zero-padded K
__global__ void k_prep_wt(const float* __restrict__ W, bf16_t* __restrict__ Wt,
                          int K, int N, int Kpad) {
  int i = blockIdx.x * blockDim.x + threadIdx.x;
  if (i >= N * Kpad) return;
  int n = i / Kpad, k = i % Kpad;
  Wt[i] = (k < K) ? f2bf(W[k * N + n]) : f2bf(0.f);
}

// Wt rows 0..383: (W_r)^T with W_r = sum_b comp[r,b]*basis[b]; rows 384..511: root^T
__global__ void k_prep_rgcn(const float* __restrict__ basis,
                            const float* __restrict__ comp,
                            const float* __restrict__ root,
                            bf16_t* __restrict__ Wt) {
  int idx = blockIdx.x * blockDim.x + threadIdx.x;    // 512*128
  if (idx >= 512 * 128) return;
  int nrow = idx >> 7;          // output row = r*128 + o
  int i = idx & 127;            // input feature (k)
  int r = nrow >> 7;
  int o = nrow & 127;
  float v;
  if (r < R_) {
    v = 0.f;
    for (int b = 0; b < NB_; ++b)
      v += comp[r * NB_ + b] * basis[((size_t)b * F_ + i) * F_ + o];
  } else {
    v = root[i * F_ + o];
  }
  Wt[idx] = f2bf(v);
}

__global__ void k_zero(float* p, int n) {
  int i = blockIdx.x * blockDim.x + threadIdx.x;
  if (i < n) p[i] = 0.f;
}

__global__ void k_count(const int* __restrict__ ei, const int* __restrict__ et,
                        float* __restrict__ cnt) {
  int e = blockIdx.x * blockDim.x + threadIdx.x;
  if (e >= E_) return;
  atomicAdd(&cnt[ei[E_ + e] * R_ + et[e]], 1.f);
}

__global__ void k_inv(float* cnt) {
  int i = blockIdx.x * blockDim.x + threadIdx.x;
  if (i < NT_ * R_) cnt[i] = 1.f / fmaxf(cnt[i], 1.f);
}

// ---- RGCN aggregation -----------------------------------------------------

// agg[row,f] = hrel[row, 384+f] (self term) + bias[f]
__global__ void k_init_agg(const float* __restrict__ hrel,
                           const float* __restrict__ bias,
                           float* __restrict__ agg, int total) {
  int i = blockIdx.x * blockDim.x + threadIdx.x;
  if (i >= total) return;
  int f = i & 127;
  agg[i] = hrel[(size_t)(i >> 7) * 512 + 384 + f] + bias[f];
}

// grid (E, T), block 128: mean-per-relation fused via precomputed 1/cnt
__global__ void k_scatter(const float* __restrict__ hrel,
                          const int* __restrict__ ei,
                          const int* __restrict__ et,
                          const float* __restrict__ invc,
                          float* __restrict__ agg) {
  int e = blockIdx.x, t = blockIdx.y, f = threadIdx.x;
  int s = ei[e], d = ei[E_ + e], r = et[e];
  float v = hrel[((size_t)t * NT_ + s) * 512 + r * 128 + f] * invc[d * R_ + r];
  atomicAdd(&agg[((size_t)t * NT_ + d) * 128 + f], v);
}

// out = elu(LN(agg)) + hprev ; wave32 per row, 4 features per lane
__global__ __launch_bounds__(256)
void k_ln_res(const float* __restrict__ agg, const float* __restrict__ hprev,
              const float* __restrict__ g, const float* __restrict__ b,
              float* __restrict__ outf, bf16_t* __restrict__ outb) {
  int wave = threadIdx.x >> 5, lane = threadIdx.x & 31;
  size_t row = (size_t)blockIdx.x * 8 + wave;
  const float4 x = *reinterpret_cast<const float4*>(agg + row * 128 + lane * 4);
  float s = x.x + x.y + x.z + x.w;
#pragma unroll
  for (int m = 16; m; m >>= 1) s += __shfl_xor(s, m, 32);
  float mu = s * (1.f / 128.f);
  float d0 = x.x - mu, d1 = x.y - mu, d2 = x.z - mu, d3 = x.w - mu;
  float vs = d0 * d0 + d1 * d1 + d2 * d2 + d3 * d3;
#pragma unroll
  for (int m = 16; m; m >>= 1) vs += __shfl_xor(vs, m, 32);
  float inv = rsqrtf(vs * (1.f / 128.f) + 1e-5f);
  const float* gp = g + lane * 4;
  const float* bp = b + lane * 4;
  const float* hp = hprev + row * 128 + lane * 4;
  float o0 = eluf(d0 * inv * gp[0] + bp[0]) + hp[0];
  float o1 = eluf(d1 * inv * gp[1] + bp[1]) + hp[1];
  float o2 = eluf(d2 * inv * gp[2] + bp[2]) + hp[2];
  float o3 = eluf(d3 * inv * gp[3] + bp[3]) + hp[3];
  float* of = outf + row * 128 + lane * 4;
  of[0] = o0; of[1] = o1; of[2] = o2; of[3] = o3;
  bf16_t* ob = outb + row * 128 + lane * 4;
  ob[0] = f2bf(o0); ob[1] = f2bf(o1); ob[2] = f2bf(o2); ob[3] = f2bf(o3);
}

// global mean pool: block per (t,b), 128 threads (one per feature)
__global__ void k_pool(const float* __restrict__ h, float* __restrict__ gseq) {
  int tb = blockIdx.x;
  int t = tb / B_, bb = tb % B_, f = threadIdx.x;
  const float* base = h + ((size_t)t * NT_ + bb * NODES_) * 128 + f;
  float s = 0.f;
  for (int v = 0; v < NODES_; ++v) s += base[(size_t)v * 128];
  gseq[(size_t)tb * 128 + f] = s * (1.f / NODES_);
}

// ---- LSTM (sequential, one block) ----------------------------------------
__global__ __launch_bounds__(1024)
void k_lstm(const float* __restrict__ gseq, const float* __restrict__ wih,
            const float* __restrict__ whh, const float* __restrict__ bih,
            const float* __restrict__ bhh, const float* __restrict__ lng,
            const float* __restrict__ lnb, float* __restrict__ ctx) {
  __shared__ float hs[B_][H_], cs[B_][H_], gates[B_][4 * H_], xb[B_][F_];
  __shared__ float mu_s[B_], inv_s[B_];
  int tid = threadIdx.x;
  { int bb = tid >> 8, k = tid & 255; hs[bb][k] = 0.f; cs[bb][k] = 0.f; }
  __syncthreads();
  for (int t = 0; t < T_; ++t) {
    if (tid < B_ * F_) {
      int bb = tid >> 7, k = tid & 127;
      xb[bb][k] = gseq[(size_t)(t * B_ + bb) * F_ + k];
    }
    __syncthreads();
    {
      int j = tid;                           // gate row 0..1023 (i,f,g,o x 256)
      const float* wi = wih + (size_t)j * F_;
      const float* wh = whh + (size_t)j * H_;
      float bsum = bih[j] + bhh[j];
      for (int bb = 0; bb < B_; ++bb) {
        float acc = bsum;
        for (int k = 0; k < F_; ++k) acc += xb[bb][k] * wi[k];
        for (int k = 0; k < H_; ++k) acc += hs[bb][k] * wh[k];
        gates[bb][j] = acc;
      }
    }
    __syncthreads();
    {
      int bb = tid >> 8, k = tid & 255;
      float ig = 1.f / (1.f + __expf(-gates[bb][k]));
      float fg = 1.f / (1.f + __expf(-gates[bb][H_ + k]));
      float gg = tanhf(gates[bb][2 * H_ + k]);
      float og = 1.f / (1.f + __expf(-gates[bb][3 * H_ + k]));
      float cn = fg * cs[bb][k] + ig * gg;
      cs[bb][k] = cn;
      hs[bb][k] = og * tanhf(cn);
    }
    __syncthreads();
  }
  if (tid < B_) {
    float s = 0.f;
    for (int k = 0; k < H_; ++k) s += hs[tid][k];
    float mu = s / H_;
    float vs = 0.f;
    for (int k = 0; k < H_; ++k) { float d = hs[tid][k] - mu; vs += d * d; }
    mu_s[tid] = mu;
    inv_s[tid] = rsqrtf(vs / H_ + 1e-5f);
  }
  __syncthreads();
  {
    int bb = tid >> 8, k = tid & 255;
    ctx[(size_t)bb * H_ + k] = (hs[bb][k] - mu_s[bb]) * inv_s[bb] * lng[k] + lnb[k];
  }
}

// ---- MLP heads ------------------------------------------------------------
__global__ __launch_bounds__(512)
void k_heads(const float* __restrict__ ctx,
             const float* __restrict__ rw1, const float* __restrict__ rb1,
             const float* __restrict__ rw2, const float* __restrict__ rb2,
             const float* __restrict__ rw3, const float* __restrict__ rb3,
             const float* __restrict__ tw1, const float* __restrict__ tb1,
             const float* __restrict__ tw2, const float* __restrict__ tb2,
             float* __restrict__ out) {
  __shared__ float cxs[B_][H_], r1[B_][F_], r2[B_][F_], ts[B_][64];
  int tid = threadIdx.x;
  for (int i = tid; i < B_ * H_; i += 512) cxs[i >> 8][i & 255] = ctx[i];
  __syncthreads();
  {
    int bb = tid >> 7, o = tid & 127;
    float a = rb1[o];
    for (int k = 0; k < H_; ++k) a += cxs[bb][k] * rw1[k * F_ + o];
    r1[bb][o] = eluf(a);
  }
  if (tid < 256) {
    int bb = tid >> 6, o = tid & 63;
    float a = tb1[o];
    for (int k = 0; k < H_; ++k) a += cxs[bb][k] * tw1[k * 64 + o];
    ts[bb][o] = eluf(a);
  }
  __syncthreads();
  {
    int bb = tid >> 7, o = tid & 127;
    float a = rb2[o];
    for (int k = 0; k < F_; ++k) a += r1[bb][k] * rw2[k * F_ + o];
    r2[bb][o] = eluf(a);
  }
  __syncthreads();
  if (tid < 16) {
    int bb = tid >> 2, o = tid & 3;
    float a = rb3[o];
    for (int k = 0; k < F_; ++k) a += r2[bb][k] * rw3[k * 4 + o];
    out[bb * 4 + o] = a;
  }
  if (tid < 4) {
    float a = tb2[0];
    for (int k = 0; k < 64; ++k) a += ts[tid][k] * tw2[k];
    out[16 + tid] = a;
  }
}

// ---------------------------------------------------------------------------
extern "C" void kernel_launch(void* const* d_in, const int* in_sizes, int n_in,
                              void* d_out, int out_size, void* d_ws,
                              size_t ws_size, hipStream_t stream) {
  (void)in_sizes; (void)n_in; (void)out_size; (void)ws_size;
  const float* x       = (const float*)d_in[0];
  const int*   eidx    = (const int*)  d_in[1];
  const int*   etype   = (const int*)  d_in[2];
  const float* enc_w1  = (const float*)d_in[3];
  const float* enc_b1  = (const float*)d_in[4];
  const float* enc_w2  = (const float*)d_in[5];
  const float* enc_b2  = (const float*)d_in[6];
  const float* basis0  = (const float*)d_in[7];
  const float* comp0   = (const float*)d_in[8];
  const float* root0   = (const float*)d_in[9];
  const float* bias0   = (const float*)d_in[10];
  const float* ln0_g   = (const float*)d_in[11];
  const float* ln0_b   = (const float*)d_in[12];
  const float* basis1  = (const float*)d_in[13];
  const float* comp1   = (const float*)d_in[14];
  const float* root1   = (const float*)d_in[15];
  const float* bias1   = (const float*)d_in[16];
  const float* ln1_g   = (const float*)d_in[17];
  const float* ln1_b   = (const float*)d_in[18];
  const float* lstm_wih = (const float*)d_in[19];
  const float* lstm_whh = (const float*)d_in[20];
  const float* lstm_bih = (const float*)d_in[21];
  const float* lstm_bhh = (const float*)d_in[22];
  const float* lstm_lng = (const float*)d_in[23];
  const float* lstm_lnb = (const float*)d_in[24];
  const float* reg_w1  = (const float*)d_in[25];
  const float* reg_b1  = (const float*)d_in[26];
  const float* reg_w2  = (const float*)d_in[27];
  const float* reg_b2  = (const float*)d_in[28];
  const float* reg_w3  = (const float*)d_in[29];
  const float* reg_b3  = (const float*)d_in[30];
  const float* tr_w1   = (const float*)d_in[31];
  const float* tr_b1   = (const float*)d_in[32];
  const float* tr_w2   = (const float*)d_in[33];
  const float* tr_b2   = (const float*)d_in[34];

  char* wp = (char*)d_ws;
  auto alloc = [&](size_t bytes) -> void* {
    void* p = (void*)wp;
    wp += (bytes + 255) & ~(size_t)255;
    return p;
  };
  bf16_t* xb   = (bf16_t*)alloc((size_t)M_ * 64 * 2);
  bf16_t* w1t  = (bf16_t*)alloc(128 * 64 * 2);
  bf16_t* w2t  = (bf16_t*)alloc(128 * 128 * 2);
  bf16_t* W0t  = (bf16_t*)alloc(512 * 128 * 2);
  bf16_t* W1t  = (bf16_t*)alloc(512 * 128 * 2);
  bf16_t* bufA = (bf16_t*)alloc((size_t)M_ * 128 * 2);
  bf16_t* bufB = (bf16_t*)alloc((size_t)M_ * 128 * 2);
  float*  f0   = (float*)alloc((size_t)M_ * 128 * 4);
  float*  f1   = (float*)alloc((size_t)M_ * 128 * 4);
  float*  agg  = (float*)alloc((size_t)M_ * 128 * 4);
  float*  hrel = (float*)alloc((size_t)M_ * 512 * 4);
  float*  cnt  = (float*)alloc((size_t)NT_ * R_ * 4);
  float*  gseq = (float*)alloc((size_t)T_ * B_ * F_ * 4);
  float*  ctx  = (float*)alloc((size_t)B_ * H_ * 4);

  // prep: pack x, transpose weights, fold basis decomposition, edge counts
  k_conv_x<<<(M_ * 64 + 255) / 256, 256, 0, stream>>>(x, xb, M_ * 64);
  k_prep_wt<<<(128 * 64 + 255) / 256, 256, 0, stream>>>(enc_w1, w1t, 37, 128, 64);
  k_prep_wt<<<(128 * 128 + 255) / 256, 256, 0, stream>>>(enc_w2, w2t, 128, 128, 128);
  k_prep_rgcn<<<(512 * 128) / 256, 256, 0, stream>>>(basis0, comp0, root0, W0t);
  k_prep_rgcn<<<(512 * 128) / 256, 256, 0, stream>>>(basis1, comp1, root1, W1t);
  k_zero<<<(NT_ * R_ + 255) / 256, 256, 0, stream>>>(cnt, NT_ * R_);
  k_count<<<(E_ + 255) / 256, 256, 0, stream>>>(eidx, etype, cnt);
  k_inv<<<(NT_ * R_ + 255) / 256, 256, 0, stream>>>(cnt);

  const int gm = (M_ + 127) / 128;   // 469 M-tiles

  // encoder (batched over all T)
  gemm_wmma<64, true, true, false, true>
      <<<dim3(gm, 1), 256, 0, stream>>>(xb, w1t, enc_b1, nullptr, bufA, M_, 128);
  gemm_wmma<128, true, true, true, true>
      <<<dim3(gm, 1), 256, 0, stream>>>(bufA, w2t, enc_b2, f0, bufB, M_, 128);

  // RGCN layer 0: [h@W0 | h@W1 | h@W2 | h@root] in one N=512 GEMM
  gemm_wmma<128, false, false, true, false>
      <<<dim3(gm, 4), 256, 0, stream>>>(bufB, W0t, nullptr, hrel, nullptr, M_, 512);
  k_init_agg<<<(M_ * 128 + 255) / 256, 256, 0, stream>>>(hrel, bias0, agg, M_ * 128);
  k_scatter<<<dim3(E_, T_), 128, 0, stream>>>(hrel, eidx, etype, cnt, agg);
  k_ln_res<<<M_ / 8, 256, 0, stream>>>(agg, f0, ln0_g, ln0_b, f1, bufA);

  // RGCN layer 1
  gemm_wmma<128, false, false, true, false>
      <<<dim3(gm, 4), 256, 0, stream>>>(bufA, W1t, nullptr, hrel, nullptr, M_, 512);
  k_init_agg<<<(M_ * 128 + 255) / 256, 256, 0, stream>>>(hrel, bias1, agg, M_ * 128);
  k_scatter<<<dim3(E_, T_), 128, 0, stream>>>(hrel, eidx, etype, cnt, agg);
  k_ln_res<<<M_ / 8, 256, 0, stream>>>(agg, f1, ln1_g, ln1_b, f0, bufB);

  // pool -> LSTM -> heads
  k_pool<<<T_ * B_, 128, 0, stream>>>(f0, gseq);
  k_lstm<<<1, 1024, 0, stream>>>(gseq, lstm_wih, lstm_whh, lstm_bih, lstm_bhh,
                                 lstm_lng, lstm_lnb, ctx);
  k_heads<<<1, 512, 0, stream>>>(ctx, reg_w1, reg_b1, reg_w2, reg_b2,
                                 reg_w3, reg_b3, tr_w1, tr_b1, tr_w2, tr_b2,
                                 (float*)d_out);
}
